// MultiHeadAttention_23450521436964
// MI455X (gfx1250) — compile-verified, matrix operand order checked by reading, not run on hardware
//
#include <hip/hip_runtime.h>

// ---------------- feature detection ----------------
#if defined(__HIP_DEVICE_COMPILE__) &&                                   \
    __has_builtin(__builtin_amdgcn_global_load_async_to_lds_b128) &&     \
    __has_builtin(__builtin_amdgcn_s_wait_asynccnt)
#define USE_ASYNC 1
#else
#define USE_ASYNC 0
#endif

// ---------------- types / helpers ----------------
typedef unsigned short u16;
typedef __attribute__((ext_vector_type(16))) __bf16 bf16x16;
typedef __attribute__((ext_vector_type(8)))  float  floatx8;
typedef __attribute__((ext_vector_type(4)))  int    int4v;

struct __align__(16) U4 { unsigned int x, y, z, w; };
union FragU { bf16x16 v; U4 q[2]; };

__device__ __forceinline__ u16 f2bf(float f) {
  unsigned int u = __float_as_uint(f);
  u += 0x7FFFu + ((u >> 16) & 1u);  // round-to-nearest-even
  return (u16)(u >> 16);
}

__device__ __forceinline__ floatx8 zero8() {
  floatx8 z = {0.f, 0.f, 0.f, 0.f, 0.f, 0.f, 0.f, 0.f};
  return z;
}

// 16B global->LDS copy: async (ASYNCcnt) on CDNA5, sync fallback otherwise.
__device__ __forceinline__ void cp_b128(const u16* g, u16* l) {
#if USE_ASYNC
  __builtin_amdgcn_global_load_async_to_lds_b128(
      (__attribute__((address_space(1))) int4v*)(void*)(u16*)g,
      (__attribute__((address_space(3))) int4v*)(void*)l, 0, 0);
#else
  *(U4*)l = *(const U4*)g;
#endif
}
__device__ __forceinline__ void async_fence() {
#if USE_ASYNC
  __builtin_amdgcn_s_wait_asynccnt(0);
#endif
}

// Load a 16x32 bf16 WMMA operand fragment (A or B side; layout symmetric).
// p0 -> element [0][0] of a 16-row tile, row stride `ld` elements (rows 16B
// aligned). CDNA5 16-bit layout: lane holds row/col (lane&15); lanes 0-15
// carry K {0..7,16..23}, lanes 16-31 carry K {8..15,24..31}, 2 elems/VGPR.
__device__ __forceinline__ bf16x16 load_frag(const u16* p0, int ld, int k0,
                                             int lane) {
  const int mn = lane & 15;
  const int half = (lane >> 4) & 1;
  const u16* p = p0 + (size_t)mn * ld + k0 + 8 * half;
  FragU f;
  f.q[0] = *(const U4*)(p);
  f.q[1] = *(const U4*)(p + 16);
  return f.v;
}

__device__ __forceinline__ floatx8 wmma_bf16(bf16x16 a, bf16x16 b, floatx8 c) {
  return __builtin_amdgcn_wmma_f32_16x16x32_bf16(false, a, false, b, (short)0, c,
                                                 false, false);
}

__device__ __forceinline__ float rmax16(float x) {
  x = fmaxf(x, __shfl_xor(x, 1));
  x = fmaxf(x, __shfl_xor(x, 2));
  x = fmaxf(x, __shfl_xor(x, 4));
  x = fmaxf(x, __shfl_xor(x, 8));
  return x;
}
__device__ __forceinline__ float rsum16(float x) {
  x += __shfl_xor(x, 1);
  x += __shfl_xor(x, 2);
  x += __shfl_xor(x, 4);
  x += __shfl_xor(x, 8);
  return x;
}

// ---------------- pack kernels (one-time fp32 -> bf16) ----------------
// Elementwise pack: x fp32 -> y bf16, 8 elems/thread (count = 8M fixed).
__global__ void __launch_bounds__(256)
pack_bf16(const float* __restrict__ x, u16* __restrict__ y) {
  const size_t base = ((size_t)blockIdx.x * 256 + threadIdx.x) * 8;
  const float4* g4 = (const float4*)(x + base);
  float4 f0 = g4[0], f1 = g4[1];
  U4 p;
  p.x = (unsigned)f2bf(f0.x) | ((unsigned)f2bf(f0.y) << 16);
  p.y = (unsigned)f2bf(f0.z) | ((unsigned)f2bf(f0.w) << 16);
  p.z = (unsigned)f2bf(f1.x) | ((unsigned)f2bf(f1.y) << 16);
  p.w = (unsigned)f2bf(f1.z) | ((unsigned)f2bf(f1.w) << 16);
  *(U4*)(y + base) = p;
}

// Transpose-pack: W fp32 [k=1024][n=1024] -> Wt bf16 [n=1024][k=1024].
// 64x64 tiles through LDS; grid (16,16) x 256 threads.
__global__ void __launch_bounds__(256)
pack_wt(const float* __restrict__ W, u16* __restrict__ Wt) {
  __shared__ u16 t[64 * 72];  // [n_local][k_local], padded
  const int tid = threadIdx.x;
  const int n0 = blockIdx.x * 64, k0 = blockIdx.y * 64;
  {
    const int r = tid >> 2;            // k_local 0..63
    const int c = (tid & 3) * 16;      // n_local base
    const float4* g4 = (const float4*)(W + (size_t)(k0 + r) * 1024 + n0 + c);
#pragma unroll
    for (int j = 0; j < 4; ++j) {
      float4 f = g4[j];
      t[(c + j * 4 + 0) * 72 + r] = f2bf(f.x);
      t[(c + j * 4 + 1) * 72 + r] = f2bf(f.y);
      t[(c + j * 4 + 2) * 72 + r] = f2bf(f.z);
      t[(c + j * 4 + 3) * 72 + r] = f2bf(f.w);
    }
  }
  __syncthreads();
  {
    const int r = tid >> 2;            // n_local 0..63
    const int c = (tid & 3) * 16;      // k_local base
    const U4* ls = (const U4*)&t[r * 72 + c];
    U4* go = (U4*)(Wt + (size_t)(n0 + r) * 1024 + k0 + c);
    go[0] = ls[0];
    go[1] = ls[1];
  }
}

// ---------------- projection GEMM ----------------
// out = (A @ W + bias) * scale.  A: bf16 [8192][1024], Wt: bf16 [n][k].
// MODE_OUT: 0 = bf16 head-split [B=4][H=16][S=2048][d=64]
//           1 = bf16 head-split transposed [B][H][d=64][S=2048]  (for V)
//           2 = fp32 row-major [8192][1024]                      (final out)
// Block 128x128, 256 thr (8 waves 4Mx2N), BK=32, async double-buffered LDS.
template <int MODE_OUT>
__global__ void __launch_bounds__(256)
gemm_proj(const u16* __restrict__ Abf, const u16* __restrict__ Wt,
          const float* __restrict__ bias, void* __restrict__ out, float scale) {
  constexpr int Kdim = 1024, N = 1024, NK = Kdim / 32;
  constexpr int LDT = 40;  // padded LDS row stride (elems); 80B, 16B aligned
  __shared__ u16 ldsA[2][128 * LDT];
  __shared__ u16 ldsB[2][128 * LDT];

  const int tid = threadIdx.x;
  const int lane = tid & 31;
  const int half = (lane >> 4) & 1;
  const int ln = lane & 15;
  const int waveId = tid >> 5;
  const int wm = waveId & 3;   // wave row (32 rows)
  const int wn = waveId >> 2;  // wave col (64 cols)
  const int m0 = blockIdx.y * 128;
  const int n0 = blockIdx.x * 128;

  const int sr = tid >> 1;           // staged row (A: m-local, B: n-local)
  const int sc = (tid & 1) * 16;     // staged col base (k-local)

  auto stage = [&](int buf, int k0) {
    const u16* ga = Abf + (size_t)(m0 + sr) * Kdim + k0 + sc;
    u16* la = &ldsA[buf][sr * LDT + sc];
    cp_b128(ga, la);
    cp_b128(ga + 8, la + 8);
    const u16* gb = Wt + (size_t)(n0 + sr) * Kdim + k0 + sc;
    u16* lb = &ldsB[buf][sr * LDT + sc];
    cp_b128(gb, lb);
    cp_b128(gb + 8, lb + 8);
  };

  floatx8 acc[2][4];
#pragma unroll
  for (int i = 0; i < 2; ++i)
#pragma unroll
    for (int j = 0; j < 4; ++j) acc[i][j] = zero8();

  stage(0, 0);
  for (int kt = 0; kt < NK; ++kt) {
    async_fence();      // this wave's tile-kt async copies have landed
    __syncthreads();    // everyone's have, and prev compute is done
    if (kt + 1 < NK) stage((kt + 1) & 1, (kt + 1) * 32);

    const u16* cA = &ldsA[kt & 1][0];
    const u16* cB = &ldsB[kt & 1][0];
    bf16x16 af[2];
#pragma unroll
    for (int mi = 0; mi < 2; ++mi)
      af[mi] = load_frag(cA + (wm * 32 + mi * 16) * LDT, LDT, 0, lane);
    bf16x16 bfr[4];
#pragma unroll
    for (int ni = 0; ni < 4; ++ni)
      bfr[ni] = load_frag(cB + (wn * 64 + ni * 16) * LDT, LDT, 0, lane);
#pragma unroll
    for (int mi = 0; mi < 2; ++mi)
#pragma unroll
      for (int ni = 0; ni < 4; ++ni)
        acc[mi][ni] = wmma_bf16(af[mi], bfr[ni], acc[mi][ni]);
  }

  // epilogue: C layout -> lane holds N=lane&15, row = v + 8*(lane>>4)
#pragma unroll
  for (int mi = 0; mi < 2; ++mi) {
#pragma unroll
    for (int ni = 0; ni < 4; ++ni) {
      const int nn = n0 + wn * 64 + ni * 16 + ln;
      const float bb = bias[nn];
#pragma unroll
      for (int v = 0; v < 8; ++v) {
        const int mm = m0 + wm * 32 + mi * 16 + v + 8 * half;
        const float val = (acc[mi][ni][v] + bb) * scale;
        if (MODE_OUT == 0) {
          const int b = mm >> 11, s = mm & 2047, hh = nn >> 6, dd = nn & 63;
          ((u16*)out)[((size_t)(b * 16 + hh) * 2048 + s) * 64 + dd] = f2bf(val);
        } else if (MODE_OUT == 1) {
          const int b = mm >> 11, s = mm & 2047, hh = nn >> 6, dd = nn & 63;
          ((u16*)out)[((size_t)(b * 16 + hh) * 64 + dd) * 2048 + s] = f2bf(val);
        } else {
          ((float*)out)[(size_t)mm * N + nn] = val;
        }
      }
    }
  }
}

// ---------------- flash attention ----------------
// Q,K: bf16 [B][H][S][64] (Q pre-scaled by 1/sqrt(64)); Vt: bf16 [B][H][64][S].
// Block = 128 thr (4 waves); block owns 64 q rows of one (b,h); wave owns 16.
// Async double-buffered K/V tiles; online softmax; P re-laid-out via LDS.
__global__ void __launch_bounds__(128)
flash_attn(const u16* __restrict__ Q, const u16* __restrict__ K,
           const u16* __restrict__ Vt, u16* __restrict__ O) {
  constexpr int S = 2048, Dh = 64, NKB = S / 64;
  const int qt = blockIdx.x;
  const int h = blockIdx.y, b = blockIdx.z;
  const int bh = b * 16 + h;
  const int tid = threadIdx.x;
  const int wave = tid >> 5;
  const int lane = tid & 31;
  const int half = (lane >> 4) & 1;
  const int ln = lane & 15;

  __shared__ u16 ldsK[2][64 * 72];   // [key s][d]
  __shared__ u16 ldsV[2][64 * 72];   // [d][key s]
  __shared__ u16 ldsP[4][16 * 72];   // wave-private P tiles

  const u16* Qb = Q + ((size_t)bh * S + qt * 64 + wave * 16) * Dh;
  bf16x16 qf[2];
  qf[0] = load_frag(Qb, Dh, 0, lane);
  qf[1] = load_frag(Qb, Dh, 32, lane);

  floatx8 oacc[4];
#pragma unroll
  for (int i = 0; i < 4; ++i) oacc[i] = zero8();
  float mrow[8], lrow[8];
#pragma unroll
  for (int v = 0; v < 8; ++v) { mrow[v] = -1e30f; lrow[v] = 0.f; }

  const u16* Kb = K + (size_t)bh * S * Dh;
  const u16* Vb = Vt + (size_t)bh * Dh * S;

  const int sr = tid >> 1;          // staged row
  const int sc = (tid & 1) * 32;    // staged col base
  auto stageKV = [&](int buf, int kb) {
    const u16* gk = Kb + (size_t)(kb * 64 + sr) * Dh + sc;
    u16* lk = &ldsK[buf][sr * 72 + sc];
    const u16* gv = Vb + (size_t)sr * S + kb * 64 + sc;
    u16* lv = &ldsV[buf][sr * 72 + sc];
#pragma unroll
    for (int j = 0; j < 4; ++j) {
      cp_b128(gk + j * 8, lk + j * 8);
      cp_b128(gv + j * 8, lv + j * 8);
    }
  };

  stageKV(0, 0);
  for (int kb = 0; kb < NKB; ++kb) {
    async_fence();
    __syncthreads();
    if (kb + 1 < NKB) stageKV((kb + 1) & 1, kb + 1);
    const u16* cK = &ldsK[kb & 1][0];
    const u16* cV = &ldsV[kb & 1][0];

    // S = Q * K^T   (16 x 64 per wave, 8 WMMAs)
    floatx8 sacc[4];
#pragma unroll
    for (int nt = 0; nt < 4; ++nt) {
      sacc[nt] = zero8();
#pragma unroll
      for (int ks = 0; ks < 2; ++ks) {
        bf16x16 kf = load_frag(cK + (nt * 16) * 72, 72, ks * 32, lane);
        sacc[nt] = wmma_bf16(qf[ks], kf, sacc[nt]);
      }
    }

    // online softmax: row for slot v is (v + 8*half); reduce over 16 lanes
    float mnew[8], alpha[8];
#pragma unroll
    for (int v = 0; v < 8; ++v) {
      float mx = fmaxf(fmaxf(sacc[0][v], sacc[1][v]),
                       fmaxf(sacc[2][v], sacc[3][v]));
      mx = rmax16(mx);
      mnew[v] = fmaxf(mrow[v], mx);
      alpha[v] = __expf(mrow[v] - mnew[v]);
    }
#pragma unroll
    for (int nt = 0; nt < 4; ++nt)
#pragma unroll
      for (int v = 0; v < 8; ++v) sacc[nt][v] = __expf(sacc[nt][v] - mnew[v]);
#pragma unroll
    for (int v = 0; v < 8; ++v) {
      float s = sacc[0][v] + sacc[1][v] + sacc[2][v] + sacc[3][v];
      s = rsum16(s);
      lrow[v] = lrow[v] * alpha[v] + s;
      mrow[v] = mnew[v];
    }
#pragma unroll
    for (int nt = 0; nt < 4; ++nt)
#pragma unroll
      for (int v = 0; v < 8; ++v) oacc[nt][v] *= alpha[v];

    // P (C-layout) -> LDS -> A-fragments (same-wave LDS is in-order)
    u16* P = &ldsP[wave][0];
#pragma unroll
    for (int nt = 0; nt < 4; ++nt)
#pragma unroll
      for (int v = 0; v < 8; ++v)
        P[(v + 8 * half) * 72 + nt * 16 + ln] = f2bf(sacc[nt][v]);

    bf16x16 pf[2];
    pf[0] = load_frag(P, 72, 0, lane);
    pf[1] = load_frag(P, 72, 32, lane);

    // O += P * V   (8 WMMAs)
#pragma unroll
    for (int nt = 0; nt < 4; ++nt)
#pragma unroll
      for (int ks = 0; ks < 2; ++ks) {
        bf16x16 vf = load_frag(cV + (nt * 16) * 72, 72, ks * 32, lane);
        oacc[nt] = wmma_bf16(pf[ks], vf, oacc[nt]);
      }
  }

  // normalize and emit bf16 [B][S][D_MODEL] for the output projection
  u16* Ob = O + ((size_t)b * S + qt * 64 + wave * 16) * 1024 + h * 64;
  float inv[8];
#pragma unroll
  for (int v = 0; v < 8; ++v) inv[v] = 1.0f / lrow[v];
#pragma unroll
  for (int nt = 0; nt < 4; ++nt)
#pragma unroll
    for (int v = 0; v < 8; ++v)
      Ob[(size_t)(v + 8 * half) * 1024 + nt * 16 + ln] =
          f2bf(oacc[nt][v] * inv[v]);
}

// ---------------- launch ----------------
extern "C" void kernel_launch(void* const* d_in, const int* in_sizes, int n_in,
                              void* d_out, int out_size, void* d_ws,
                              size_t ws_size, hipStream_t stream) {
  (void)in_sizes; (void)n_in; (void)out_size; (void)ws_size;
  const float* query = (const float*)d_in[0];
  const float* key   = (const float*)d_in[1];
  const float* value = (const float*)d_in[2];
  const float* Wq = (const float*)d_in[3];
  const float* bq = (const float*)d_in[4];
  const float* Wk = (const float*)d_in[5];
  const float* bk = (const float*)d_in[6];
  const float* Wv = (const float*)d_in[7];
  const float* bv = (const float*)d_in[8];
  const float* Wo = (const float*)d_in[9];
  const float* bo = (const float*)d_in[10];

  // workspace: Qb,Kb,Vb,Ab (bf16 8192x1024 each = 64MB) + 4 packed weights
  // (bf16 1024x1024 each = 8MB). Ab doubles as the input-pack buffer.
  const size_t elems = (size_t)8192 * 1024;
  const size_t welems = (size_t)1024 * 1024;
  u16* Qb = (u16*)d_ws;
  u16* Kb = Qb + elems;
  u16* Vb = Kb + elems;
  u16* Ab = Vb + elems;
  u16* Wtq = Ab + elems;
  u16* Wtk = Wtq + welems;
  u16* Wtv = Wtk + welems;
  u16* Wto = Wtv + welems;

  dim3 wg(16, 16);
  pack_wt<<<wg, 256, 0, stream>>>(Wq, Wtq);
  pack_wt<<<wg, 256, 0, stream>>>(Wk, Wtk);
  pack_wt<<<wg, 256, 0, stream>>>(Wv, Wtv);
  pack_wt<<<wg, 256, 0, stream>>>(Wo, Wto);

  dim3 gg(8, 64), bg(256);
  // Q gets the 1/sqrt(depth)=0.125 attention scale folded into the projection
  pack_bf16<<<4096, 256, 0, stream>>>(query, Ab);
  gemm_proj<0><<<gg, bg, 0, stream>>>(Ab, Wtq, bq, Qb, 0.125f);
  pack_bf16<<<4096, 256, 0, stream>>>(key, Ab);
  gemm_proj<0><<<gg, bg, 0, stream>>>(Ab, Wtk, bk, Kb, 1.0f);
  pack_bf16<<<4096, 256, 0, stream>>>(value, Ab);
  gemm_proj<1><<<gg, bg, 0, stream>>>(Ab, Wtv, bv, Vb, 1.0f);

  flash_attn<<<dim3(32, 16, 4), 128, 0, stream>>>(Qb, Kb, Vb, Ab);

  gemm_proj<2><<<gg, bg, 0, stream>>>(Ab, Wto, bo, d_out, 1.0f);
}
